// MixtureOfExperts_79164837200270
// MI455X (gfx1250) — matrix operand order repaired and checked by reading, not compiled, not run on hardware
//
#include <hip/hip_runtime.h>
#include <hip/hip_bf16.h>
#include <math.h>

// ---------------------------------------------------------------------------
// MoE forward for MI455X (gfx1250, wave32, WMMA 16x16x32 bf16)
// B*L = 8192 tokens, D=1024, F=4096, E=8, top-2, capacity = 1280
// ---------------------------------------------------------------------------

typedef __attribute__((ext_vector_type(8)))  float  v8f;
typedef __attribute__((ext_vector_type(8)))  __bf16 v8bf;
typedef __attribute__((ext_vector_type(16))) __bf16 v16bf;
typedef int v4i __attribute__((vector_size(16)));   // matches builtin param type

#define N_TOK  8192
#define DMODEL 1024
#define FDIM   4096
#define NEXP   8
#define TOPK   2
#define CAP    1280
#define NA     (N_TOK * TOPK)   // 16384 assignments

// ---- CDNA5 async global->LDS staging (guarded; falls back to plain loads) --
#if __has_builtin(__builtin_amdgcn_global_load_async_to_lds_b128) && \
    __has_builtin(__builtin_amdgcn_s_wait_asynccnt)
#define HAVE_ASYNC_LDS 1
#else
#define HAVE_ASYNC_LDS 0
#endif

__device__ __forceinline__ void stage16(const void* g, void* l) {
#if HAVE_ASYNC_LDS
    __builtin_amdgcn_global_load_async_to_lds_b128((v4i*)g, (v4i*)l, 0, 0);
#else
    *(v8bf*)l = *(const v8bf*)g;
#endif
}
__device__ __forceinline__ void stage_wait() {
#if HAVE_ASYNC_LDS
    __builtin_amdgcn_s_wait_asynccnt(0);
#endif
}

__device__ __forceinline__ unsigned pack_bf16(float a, float b) {
    union { __bf16 h[2]; unsigned u; } p;
    p.h[0] = (__bf16)a;
    p.h[1] = (__bf16)b;
    return p.u;
}

// ---------------------------------------------------------------------------
// 1) Router: logits = x @ rw^T, softmax over 8 experts, top-2, normalize.
// ---------------------------------------------------------------------------
__global__ __launch_bounds__(256) void router_topk(
    const float* __restrict__ x, const float* __restrict__ rw,
    int* __restrict__ topi, float* __restrict__ topw)
{
    const int wave = threadIdx.x >> 5;
    const int lane = threadIdx.x & 31;
    const int t = blockIdx.x * 8 + wave;
    const float* xr = x + (size_t)t * DMODEL;

    float acc[NEXP];
#pragma unroll
    for (int e = 0; e < NEXP; e++) acc[e] = 0.f;

    for (int d = lane; d < DMODEL; d += 32) {
        float xv = xr[d];
#pragma unroll
        for (int e = 0; e < NEXP; e++) acc[e] += xv * rw[e * DMODEL + d];
    }
#pragma unroll
    for (int e = 0; e < NEXP; e++) {
#pragma unroll
        for (int off = 16; off; off >>= 1) acc[e] += __shfl_xor(acc[e], off, 32);
    }
    if (lane == 0) {
        float m = acc[0];
#pragma unroll
        for (int e = 1; e < NEXP; e++) m = fmaxf(m, acc[e]);
        float p[NEXP], s = 0.f;
#pragma unroll
        for (int e = 0; e < NEXP; e++) { p[e] = __expf(acc[e] - m); s += p[e]; }
        float inv = 1.f / s;
        int i0 = 0; float v0 = -1.f;
#pragma unroll
        for (int e = 0; e < NEXP; e++) { float pe = p[e] * inv; p[e] = pe; if (pe > v0) { v0 = pe; i0 = e; } }
        int i1 = 0; float v1 = -1.f;
#pragma unroll
        for (int e = 0; e < NEXP; e++) { if (e != i0 && p[e] > v1) { v1 = p[e]; i1 = e; } }
        float z = 1.f / (v0 + v1 + 1e-6f);
        topi[t * 2 + 0] = i0;  topw[t * 2 + 0] = v0 * z;
        topi[t * 2 + 1] = i1;  topw[t * 2 + 1] = v1 * z;
    }
}

// ---------------------------------------------------------------------------
// 2) Stable per-expert rank by descending weight == lexsort((-fw, fe)).
//    Monotone key: (expert << 29) | (float_bits(w) >> 1); w in (0,1).
// ---------------------------------------------------------------------------
__device__ __forceinline__ unsigned key_of(int e, float w) {
    unsigned b = __float_as_uint(w);
    return ((unsigned)e << 29) | (b >> 1);
}

__global__ __launch_bounds__(256) void rank_assign(
    const int* __restrict__ topi, const float* __restrict__ topw,
    int* __restrict__ apos, int* __restrict__ smap)
{
    __shared__ unsigned skey[8192];
    const int i = blockIdx.x * 256 + threadIdx.x;
    const int e = topi[i];
    const unsigned ki = key_of(e, topw[i]);
    int rank = 0;
    for (int h = 0; h < 2; h++) {
        __syncthreads();
        for (int j = threadIdx.x; j < 8192; j += 256) {
            int g = h * 8192 + j;
            skey[j] = key_of(topi[g], topw[g]);
        }
        __syncthreads();
        const int base = h * 8192;
        for (int j = 0; j < 8192; j++) {
            unsigned kj = skey[j];
            if (((kj ^ ki) >> 29) == 0) {                 // same expert
                if (kj > ki || (kj == ki && (base + j) < i)) rank++;
            }
        }
    }
    if (rank < CAP) {
        apos[i] = rank;
        smap[e * CAP + rank] = i >> 1;   // token id (i = t*K + slot)
    } else {
        apos[i] = -1;                    // dropped by capacity
    }
}

__global__ void init_smap(int* __restrict__ smap) {
    int i = blockIdx.x * 256 + threadIdx.x;
    if (i < NEXP * CAP) smap[i] = -1;
}

// ---------------------------------------------------------------------------
// 3) Dispatch: disp[e][c][d] (bf16) = x[token] or zeros for unfilled slots.
// ---------------------------------------------------------------------------
__global__ __launch_bounds__(256) void build_disp(
    const float* __restrict__ x, const int* __restrict__ smap,
    __bf16* __restrict__ disp)
{
    const int row = blockIdx.x;              // e*CAP + c
    const int t = smap[row];
    __bf16* dst = disp + (size_t)row * DMODEL;
    if (t >= 0) {
        const float* src = x + (size_t)t * DMODEL;
        for (int j = threadIdx.x; j < DMODEL; j += 256) dst[j] = (__bf16)src[j];
    } else {
        for (int j = threadIdx.x; j < DMODEL; j += 256) dst[j] = (__bf16)0.f;
    }
}

// ---------------------------------------------------------------------------
// 4) Fused gate/up GEMM + SiLU: h = silu(disp@wg) * (disp@wu), bf16 out.
//    Block 256 thr (8 waves), tile 64(M) x 128(N), K-step 64 (2 WMMA k-steps).
//    Wave tile 32x32 => 2x2 WMMA positions x {g,u} x 2 k-steps = 16 WMMA/iter.
//    Fragment lane layouts used here (wave32):
//      A 16x32 bf16 : lane holds row M=lane%16, K = 16*lh..+15 (contiguous)
//      B 32x16 bf16 : lane holds col N=lane%16, K = 16*lh..+15 (contiguous in
//                     the transposed [n][k] LDS image)
//      C/D 16x16 f32: elem r -> (m = r + 8*lh, n = lane%16)
// ---------------------------------------------------------------------------
__global__ __launch_bounds__(256) void gemm_gate_up(
    const __bf16* __restrict__ disp, const float* __restrict__ wg,
    const float* __restrict__ wu, __bf16* __restrict__ h)
{
    __shared__ __align__(64) __bf16 sA[64 * 64];     // 8 KB
    __shared__ __align__(64) __bf16 sBg[128 * 64];   // 16 KB
    __shared__ __align__(64) __bf16 sBu[128 * 64];   // 16 KB

    const int e = blockIdx.z;
    const int cBase = blockIdx.y * 64;
    const int fBase = blockIdx.x * 128;
    const int tid = threadIdx.x;
    const int lane = tid & 31, wave = tid >> 5;
    const int wm = wave & 1, wn = wave >> 1;         // 2 x 4 wave grid
    const int lr = lane & 15, lh = lane >> 4;

    const __bf16* A  = disp + ((size_t)e * CAP + cBase) * DMODEL;
    const float*  Bg = wg + (size_t)e * DMODEL * FDIM + fBase;
    const float*  Bu = wu + (size_t)e * DMODEL * FDIM + fBase;

    v8f accg[2][2] = {};
    v8f accu[2][2] = {};

    const int am  = tid >> 2,  ak = (tid & 3) * 16;  // A tile: 64 rows x 64 cols
    const int bkp = (tid & 31) * 2;                  // B: even k of a k-pair
    const int bn  = (tid >> 5) * 16;                 // B: 16-column group

    for (int k0 = 0; k0 < DMODEL; k0 += 64) {
        // ---- stage A: 32 B per thread via async global->LDS copies ----
        stage16(A + (size_t)am * DMODEL + k0 + ak,     &sA[am * 64 + ak]);
        stage16(A + (size_t)am * DMODEL + k0 + ak + 8, &sA[am * 64 + ak + 8]);

        // ---- stage B: two f32 rows (k, k+1), pack bf16 pairs, [n][k] ----
        const float* pg0 = Bg + (size_t)(k0 + bkp) * FDIM + bn;
        const float* pu0 = Bu + (size_t)(k0 + bkp) * FDIM + bn;
        const float4* qg0 = (const float4*)pg0;
        const float4* qg1 = (const float4*)(pg0 + FDIM);
        const float4* qu0 = (const float4*)pu0;
        const float4* qu1 = (const float4*)(pu0 + FDIM);
#pragma unroll
        for (int jj = 0; jj < 4; jj++) {
            float4 g0 = qg0[jj], g1 = qg1[jj];
            float4 u0 = qu0[jj], u1 = qu1[jj];
            int n = bn + jj * 4;
            *(unsigned*)&sBg[(n + 0) * 64 + bkp] = pack_bf16(g0.x, g1.x);
            *(unsigned*)&sBg[(n + 1) * 64 + bkp] = pack_bf16(g0.y, g1.y);
            *(unsigned*)&sBg[(n + 2) * 64 + bkp] = pack_bf16(g0.z, g1.z);
            *(unsigned*)&sBg[(n + 3) * 64 + bkp] = pack_bf16(g0.w, g1.w);
            *(unsigned*)&sBu[(n + 0) * 64 + bkp] = pack_bf16(u0.x, u1.x);
            *(unsigned*)&sBu[(n + 1) * 64 + bkp] = pack_bf16(u0.y, u1.y);
            *(unsigned*)&sBu[(n + 2) * 64 + bkp] = pack_bf16(u0.z, u1.z);
            *(unsigned*)&sBu[(n + 3) * 64 + bkp] = pack_bf16(u0.w, u1.w);
        }
        if (k0 + 64 < DMODEL) {                       // prefetch next K-slab
            __builtin_prefetch(pg0 + (size_t)64 * FDIM, 0, 0);
            __builtin_prefetch(pu0 + (size_t)64 * FDIM, 0, 0);
        }
        stage_wait();
        __syncthreads();

        v16bf af[2][2];
#pragma unroll
        for (int s = 0; s < 2; s++) {
#pragma unroll
            for (int mi = 0; mi < 2; mi++) {
                int m = wm * 32 + mi * 16 + lr;
                v8bf lo = *(const v8bf*)&sA[m * 64 + s * 32 + lh * 16];
                v8bf hi = *(const v8bf*)&sA[m * 64 + s * 32 + lh * 16 + 8];
                af[s][mi] = __builtin_shufflevector(lo, hi, 0,1,2,3,4,5,6,7,8,9,10,11,12,13,14,15);
            }
        }
        v16bf bg[2][2], bu[2][2];
#pragma unroll
        for (int s = 0; s < 2; s++) {
#pragma unroll
            for (int ni = 0; ni < 2; ni++) {
                int n = wn * 32 + ni * 16 + lr;
                bg[s][ni] = *(const v16bf*)&sBg[n * 64 + s * 32 + lh * 16];
                bu[s][ni] = *(const v16bf*)&sBu[n * 64 + s * 32 + lh * 16];
            }
        }
#pragma unroll
        for (int s = 0; s < 2; s++) {
#pragma unroll
            for (int mi = 0; mi < 2; mi++) {
#pragma unroll
                for (int ni = 0; ni < 2; ni++) {
                    accg[mi][ni] = __builtin_amdgcn_wmma_f32_16x16x32_bf16(
                        false, af[s][mi], false, bg[s][ni], (short)0, accg[mi][ni], false, false);
                    accu[mi][ni] = __builtin_amdgcn_wmma_f32_16x16x32_bf16(
                        false, af[s][mi], false, bu[s][ni], (short)0, accu[mi][ni], false, false);
                }
            }
        }
        __syncthreads();
    }

    // Epilogue: h = silu(g) * u = g * rcp(1 + exp(-g)) * u, stored bf16
#pragma unroll
    for (int mi = 0; mi < 2; mi++) {
#pragma unroll
        for (int ni = 0; ni < 2; ni++) {
#pragma unroll
            for (int r = 0; r < 8; r++) {
                float gv = accg[mi][ni][r];
                float uv = accu[mi][ni][r];
                float sg = __builtin_amdgcn_rcpf(1.f + __expf(-gv));
                float hv = gv * sg * uv;
                int c = cBase + wm * 32 + mi * 16 + (r + lh * 8);
                int f = fBase + wn * 32 + ni * 16 + lr;
                h[((size_t)e * CAP + c) * FDIM + f] = (__bf16)hv;
            }
        }
    }
}

// ---------------------------------------------------------------------------
// 5) Down projection: eo = h @ wd (per expert), f32 out. Same tiling, K=4096.
// ---------------------------------------------------------------------------
__global__ __launch_bounds__(256) void gemm_down(
    const __bf16* __restrict__ hbuf, const float* __restrict__ wd,
    float* __restrict__ eo)
{
    __shared__ __align__(64) __bf16 sA[64 * 64];     // 8 KB
    __shared__ __align__(64) __bf16 sB[128 * 64];    // 16 KB

    const int e = blockIdx.z;
    const int cBase = blockIdx.y * 64;
    const int dBase = blockIdx.x * 128;
    const int tid = threadIdx.x;
    const int lane = tid & 31, wave = tid >> 5;
    const int wm = wave & 1, wn = wave >> 1;
    const int lr = lane & 15, lh = lane >> 4;

    const __bf16* A = hbuf + ((size_t)e * CAP + cBase) * FDIM;
    const float*  B = wd + (size_t)e * FDIM * DMODEL + dBase;

    v8f acc[2][2] = {};

    const int am  = tid >> 2,  ak = (tid & 3) * 16;
    const int bkp = (tid & 31) * 2;
    const int bn  = (tid >> 5) * 16;

    for (int k0 = 0; k0 < FDIM; k0 += 64) {
        stage16(A + (size_t)am * FDIM + k0 + ak,     &sA[am * 64 + ak]);
        stage16(A + (size_t)am * FDIM + k0 + ak + 8, &sA[am * 64 + ak + 8]);

        const float* pb0 = B + (size_t)(k0 + bkp) * DMODEL + bn;
        const float4* q0 = (const float4*)pb0;
        const float4* q1 = (const float4*)(pb0 + DMODEL);
#pragma unroll
        for (int jj = 0; jj < 4; jj++) {
            float4 a0 = q0[jj], a1 = q1[jj];
            int n = bn + jj * 4;
            *(unsigned*)&sB[(n + 0) * 64 + bkp] = pack_bf16(a0.x, a1.x);
            *(unsigned*)&sB[(n + 1) * 64 + bkp] = pack_bf16(a0.y, a1.y);
            *(unsigned*)&sB[(n + 2) * 64 + bkp] = pack_bf16(a0.z, a1.z);
            *(unsigned*)&sB[(n + 3) * 64 + bkp] = pack_bf16(a0.w, a1.w);
        }
        if (k0 + 64 < FDIM)
            __builtin_prefetch(pb0 + (size_t)64 * DMODEL, 0, 0);
        stage_wait();
        __syncthreads();

        v16bf af[2][2], bf[2][2];
#pragma unroll
        for (int s = 0; s < 2; s++) {
#pragma unroll
            for (int mi = 0; mi < 2; mi++) {
                int m = wm * 32 + mi * 16 + lr;
                v8bf lo = *(const v8bf*)&sA[m * 64 + s * 32 + lh * 16];
                v8bf hi = *(const v8bf*)&sA[m * 64 + s * 32 + lh * 16 + 8];
                af[s][mi] = __builtin_shufflevector(lo, hi, 0,1,2,3,4,5,6,7,8,9,10,11,12,13,14,15);
            }
#pragma unroll
            for (int ni = 0; ni < 2; ni++) {
                int n = wn * 32 + ni * 16 + lr;
                bf[s][ni] = *(const v16bf*)&sB[n * 64 + s * 32 + lh * 16];
            }
        }
#pragma unroll
        for (int s = 0; s < 2; s++) {
#pragma unroll
            for (int mi = 0; mi < 2; mi++) {
#pragma unroll
                for (int ni = 0; ni < 2; ni++) {
                    acc[mi][ni] = __builtin_amdgcn_wmma_f32_16x16x32_bf16(
                        false, af[s][mi], false, bf[s][ni], (short)0, acc[mi][ni], false, false);
                }
            }
        }
        __syncthreads();
    }

#pragma unroll
    for (int mi = 0; mi < 2; mi++) {
#pragma unroll
        for (int ni = 0; ni < 2; ni++) {
#pragma unroll
            for (int r = 0; r < 8; r++) {
                int c = cBase + wm * 32 + mi * 16 + (r + lh * 8);
                int d = dBase + wn * 32 + ni * 16 + lr;
                eo[((size_t)e * CAP + c) * DMODEL + d] = acc[mi][ni][r];
            }
        }
    }
}

// ---------------------------------------------------------------------------
// 6) Combine: out[t] = sum over the token's kept slots of w * eo[e][pos].
// ---------------------------------------------------------------------------
__global__ __launch_bounds__(256) void combine(
    const float* __restrict__ eo, const int* __restrict__ topi,
    const float* __restrict__ topw, const int* __restrict__ apos,
    float* __restrict__ out)
{
    const int t = blockIdx.x;
    const int e0 = topi[2 * t], e1 = topi[2 * t + 1];
    const float w0 = topw[2 * t], w1 = topw[2 * t + 1];
    const int p0 = apos[2 * t], p1 = apos[2 * t + 1];
    const float* r0 = (p0 >= 0) ? eo + ((size_t)(e0 * CAP + p0)) * DMODEL : nullptr;
    const float* r1 = (p1 >= 0) ? eo + ((size_t)(e1 * CAP + p1)) * DMODEL : nullptr;
    for (int d = threadIdx.x; d < DMODEL; d += 256) {
        float a = 0.f;
        if (r0) a += w0 * r0[d];
        if (r1) a += w1 * r1[d];
        out[(size_t)t * DMODEL + d] = a;
    }
}

// ---------------------------------------------------------------------------
// Host launcher
// ---------------------------------------------------------------------------
extern "C" void kernel_launch(void* const* d_in, const int* in_sizes, int n_in,
                              void* d_out, int out_size, void* d_ws, size_t ws_size,
                              hipStream_t stream)
{
    (void)in_sizes; (void)n_in; (void)out_size; (void)ws_size;
    const float* x  = (const float*)d_in[0];
    const float* rw = (const float*)d_in[1];
    const float* wg = (const float*)d_in[2];
    const float* wu = (const float*)d_in[3];
    const float* wd = (const float*)d_in[4];
    float* out = (float*)d_out;

    char* ws = (char*)d_ws;
    size_t off = 0;
    auto take = [&](size_t bytes) -> char* {
        char* p = ws + off;
        off = (off + bytes + 255) & ~(size_t)255;
        return p;
    };
    int*    topi = (int*)take((size_t)NA * sizeof(int));
    float*  topw = (float*)take((size_t)NA * sizeof(float));
    int*    apos = (int*)take((size_t)NA * sizeof(int));
    int*    smap = (int*)take((size_t)NEXP * CAP * sizeof(int));
    __bf16* disp = (__bf16*)take((size_t)NEXP * CAP * DMODEL * sizeof(__bf16));
    __bf16* hbuf = (__bf16*)take((size_t)NEXP * CAP * FDIM * sizeof(__bf16));
    float*  eo   = (float*)take((size_t)NEXP * CAP * DMODEL * sizeof(float));

    init_smap<<<dim3((NEXP * CAP + 255) / 256), dim3(256), 0, stream>>>(smap);
    router_topk<<<dim3(N_TOK / 8), dim3(256), 0, stream>>>(x, rw, topi, topw);
    rank_assign<<<dim3(NA / 256), dim3(256), 0, stream>>>(topi, topw, apos, smap);
    build_disp<<<dim3(NEXP * CAP), dim3(256), 0, stream>>>(x, smap, disp);
    gemm_gate_up<<<dim3(FDIM / 128, CAP / 64, NEXP), dim3(256), 0, stream>>>(disp, wg, wu, hbuf);
    gemm_down<<<dim3(DMODEL / 128, CAP / 64, NEXP), dim3(256), 0, stream>>>(hbuf, wd, eo);
    combine<<<dim3(N_TOK), dim3(256), 0, stream>>>(eo, topi, topw, apos, out);
}
